// SwinTransformerBlock_35502199668792
// MI455X (gfx1250) — matrix operand order repaired and checked by reading, not compiled, not run on hardware
//
#include <hip/hip_runtime.h>
#include <hip/hip_bf16.h>

#define WSZ   7
#define SS    3
#define MM    49
#define DIM   192
#define HEADS 6
#define HD    32
#define HID   768
#define IMG   56
#define NWIN  4096          // 64 batch * 8*8 windows
#define SCALE 0.17677669529663687f
#define NEGV  (-1e9f)

typedef _Float16 h8   __attribute__((ext_vector_type(8)));
typedef _Float16 v16h __attribute__((ext_vector_type(16)));
typedef float    v8f  __attribute__((ext_vector_type(8)));
typedef unsigned int v4u __attribute__((ext_vector_type(4)));
typedef int      v4i  __attribute__((ext_vector_type(4)));
typedef int      v8i  __attribute__((ext_vector_type(8)));

__device__ __forceinline__ v16h mkfrag(h8 lo, h8 hi) {
    return __builtin_shufflevector(lo, hi, 0,1,2,3,4,5,6,7,8,9,10,11,12,13,14,15);
}
// A fragment (16x32 f16): lane -> M row; K elems = {g*16 + h*8 + j}, g in {0,1}, j in 0..7
__device__ __forceinline__ v16h frag_a(const _Float16* rowk, int h) {
    h8 lo = *(const h8*)(rowk + h * 8);
    h8 hi = *(const h8*)(rowk + 16 + h * 8);
    return mkfrag(lo, hi);
}
// B fragment (32x16 f16): lane -> N col; K elems = {g*16 + h*8 + j}, g in {0,1}, j in 0..7
__device__ __forceinline__ v16h frag_b(const _Float16* rowk, int h) {
    const h8* p = (const h8*)(rowk + h * 8);
    return mkfrag(p[0], p[2]);
}
__device__ __forceinline__ v8f wmma32(v16h a, v16h b, v8f c) {
    return __builtin_amdgcn_wmma_f32_16x16x32_f16(false, a, false, b, (short)0, c, false, false);
}
__device__ __forceinline__ v8f vzero8() {
    v8f z; for (int j = 0; j < 8; j++) z[j] = 0.0f; return z;
}

// ---- Tensor Data Mover: 1-D contiguous copy global -> LDS (n_elem2B f16 elements) ----
// D# per CDNA5 ISA ch.8: group0 = {flags/count, lds_addr, gaddr_lo, gaddr_hi|type},
// group1 = dims/strides. Issued once per executing wave; tracked by TENSORcnt.
__device__ __forceinline__ void tdm_load_lds_1d(void* lds_dst, const void* gsrc, int n_elem2B) {
    unsigned long long ga = (unsigned long long)(size_t)gsrc;
    unsigned lds = (unsigned)(size_t)lds_dst;   // low 32 bits of generic ptr = LDS offset
    v4u g0;
    g0[0] = 1u;                                          // count=1 (valid user descriptor)
    g0[1] = lds;                                         // lds_addr (bytes)
    g0[2] = (unsigned)ga;                                // global_addr[31:0]
    g0[3] = (unsigned)((ga >> 32) & 0x01FFFFFFu) | (2u << 30);  // global_addr[56:32], type=2
    unsigned n = (unsigned)n_elem2B;                     // fits in 16 bits (1568)
    v8i g1;
    g1[0] = (int)(1u << 16);      // workgroup_mask=0, data_size=1 (2 bytes)
    g1[1] = (int)(n << 16);       // tensor_dim0[15:0]
    g1[2] = (int)(1u << 16);      // tensor_dim0[31:16]=0 | tensor_dim1=1
    g1[3] = (int)(n << 16);       // tensor_dim1 hi=0 | tile_dim0 = n
    g1[4] = 1;                    // tile_dim1=1, tile_dim2=0
    g1[5] = (int)n;               // tensor_dim0_stride[31:0]
    g1[6] = (int)(n << 16);       // stride0[47:32]=0 | tensor_dim1_stride[15:0]
    g1[7] = 0;                    // tensor_dim1_stride[47:16]=0
    v4i z; z[0] = z[1] = z[2] = z[3] = 0;
    asm volatile("tensor_load_to_lds %0, %1, %2, %3"
                 :: "s"(g0), "s"(g1), "s"(z), "s"(z)
                 : "memory");
}

// ---------------- K0: convert weights to f16 ----------------
__global__ void k0_convert(const float* __restrict__ qkvw, const float* __restrict__ outw,
                           const float* __restrict__ w1,   const float* __restrict__ w2,
                           _Float16* qkvw16, _Float16* outw16, _Float16* w116, _Float16* w216) {
    int i = blockIdx.x * blockDim.x + threadIdx.x;
    int stride = gridDim.x * blockDim.x;
    for (int e = i; e < 576 * DIM; e += stride) qkvw16[e] = (_Float16)qkvw[e];
    for (int e = i; e < DIM * DIM; e += stride) outw16[e] = (_Float16)outw[e];
    for (int e = i; e < HID * DIM; e += stride) w116[e]   = (_Float16)w1[e];
    for (int e = i; e < DIM * HID; e += stride) w216[e]   = (_Float16)w2[e];
}

// ---------------- K1: window gather + LN1 + QKV GEMM ----------------
__global__ __launch_bounds__(256) void k1_qkv(
    const float* __restrict__ x, const float* __restrict__ ln1_g, const float* __restrict__ ln1_b,
    const _Float16* __restrict__ qkvw16,
    float* __restrict__ xw, _Float16* __restrict__ qws, _Float16* __restrict__ kws,
    _Float16* __restrict__ vws) {
    __shared__ __align__(16) float    xs[MM * DIM];  // 37632 B
    __shared__ __align__(16) _Float16 y[64 * DIM];   // 24576 B
    int win = blockIdx.x;
    int b = win >> 6, wi = win & 63, wh = wi >> 3, ww = wi & 7;
    int tid = threadIdx.x;

    // gather shifted window (roll -SS => src = dst + SS mod 56); also write residual xw
    for (int e = tid; e < MM * DIM; e += 256) {
        int t = e % MM, c = e / MM;
        int p = t / WSZ, q = t % WSZ;
        int sh = (wh * WSZ + p + SS) % IMG;
        int sw = (ww * WSZ + q + SS) % IMG;
        float val = x[(((long)b * DIM + c) * IMG + sh) * IMG + sw];
        xs[t * DIM + c] = val;
        xw[((long)win * MM + t) * DIM + c] = val;
    }
    __syncthreads();

    // LayerNorm per token -> y (f16), pad rows 49..63 = 0
    if (tid < 64) {
        if (tid < MM) {
            float s = 0.f, s2 = 0.f;
            for (int c = 0; c < DIM; c++) { float u = xs[tid * DIM + c]; s += u; s2 += u * u; }
            float m = s / DIM;
            float var = s2 / DIM - m * m;
            float inv = rsqrtf(var + 1e-5f);
            for (int c = 0; c < DIM; c++)
                y[tid * DIM + c] = (_Float16)((xs[tid * DIM + c] - m) * inv * ln1_g[c] + ln1_b[c]);
        } else {
            for (int c = 0; c < DIM; c++) y[tid * DIM + c] = (_Float16)0.0f;
        }
    }
    __syncthreads();

    // QKV GEMM: [64 x 576] = y[64x192] * qkv_w^T ; 4x36 = 144 tiles over 8 waves
    int wv = tid >> 5, lane = tid & 31, ln16 = lane & 15, h = lane >> 4;
    for (int i = 0; i < 18; i++) {
        int id = wv * 18 + i;
        int mb = (id / 36) * 16, nb = (id % 36) * 16;
        if (i + 1 < 18) {   // warm L0/L2 for next tile's B rows (global_prefetch_b8)
            int nb2 = (((id + 1) % 36) * 16);
            __builtin_prefetch(&qkvw16[(nb2 + ln16) * DIM], 0, 1);
        }
        v8f acc = vzero8();
        for (int ks = 0; ks < 6; ks++) {
            v16h a = frag_a(&y[(mb + ln16) * DIM + ks * 32], h);
            v16h bf = frag_b(&qkvw16[(nb + ln16) * DIM + ks * 32], h);
            acc = wmma32(a, bf, acc);
        }
        int n = nb + ln16;
        int s = n / DIM, r = n % DIM, head = r / HD, d = r % HD;
        _Float16* dst = (s == 0) ? qws : ((s == 1) ? kws : vws);
        float scl = (s == 0) ? SCALE : 1.0f;
        long base = ((long)(win * HEADS + head)) * MM * HD + d;
        for (int j = 0; j < 8; j++) {
            int m = mb + j + 8 * h;
            if (m < MM) dst[base + (long)m * HD] = (_Float16)(acc[j] * scl);
        }
    }
}

// ---------------- K2: attention (bias+mask+softmax) + out-proj + residual ----------------
__global__ __launch_bounds__(256) void k2_attn(
    const _Float16* __restrict__ qws, const _Float16* __restrict__ kws,
    const _Float16* __restrict__ vws, const _Float16* __restrict__ outw16,
    const float* __restrict__ out_b, const float* __restrict__ rel_table,
    float* __restrict__ xw) {
    __shared__ __align__(16) _Float16 qs[64 * HD];   // 4096 B
    __shared__ __align__(16) _Float16 ks[64 * HD];   // 4096 B
    __shared__ __align__(16) _Float16 vt[HD * 64];   // 4096 B (transposed V)
    __shared__ __align__(16) float    lg[64 * 64];   // 16384 B
    __shared__ __align__(16) _Float16 ps[64 * 64];   // 8192 B
    __shared__ __align__(16) _Float16 os[64 * DIM];  // 24576 B
    __shared__ float btab[169];
    __shared__ int   cnt[MM];

    int win = blockIdx.x;
    int wi = win & 63, wh = wi >> 3, ww = wi & 7;
    int tid = threadIdx.x, wv = tid >> 5, lane = tid & 31, ln16 = lane & 15, h = lane >> 4;

    if (tid < MM) {  // shift-mask region label per token
        int gh = wh * WSZ + tid / WSZ;
        int gw = ww * WSZ + tid % WSZ;
        int lh = (gh < 49) ? 0 : ((gh < 53) ? 1 : 2);
        int lw = (gw < 49) ? 0 : ((gw < 53) ? 1 : 2);
        cnt[tid] = lh * 3 + lw;
    }
    // zero pad rows (49..63) of qs/ks and pad cols of vt once; TDM only writes rows 0..48
    for (int e = tid; e < 64 * HD; e += 256) {
        int r = e >> 5, d = e & 31;
        if (r >= MM) { qs[e] = (_Float16)0.0f; ks[e] = (_Float16)0.0f; vt[d * 64 + r] = (_Float16)0.0f; }
    }

    for (int head = 0; head < HEADS; head++) {
        long base = ((long)(win * HEADS + head)) * MM * HD;
        // async TDM staging of Q and K tiles (contiguous 49*32 f16) into LDS
        if (wv == 0)      tdm_load_lds_1d(qs, qws + base, MM * HD);
        else if (wv == 1) tdm_load_lds_1d(ks, kws + base, MM * HD);
        // V needs an LDS-side transpose -> manual staging
        for (int e = tid; e < MM * HD; e += 256) {
            int r = e >> 5, d = e & 31;
            vt[d * 64 + r] = vws[base + r * HD + d];
        }
        if (tid < 169) btab[tid] = rel_table[tid * HEADS + head];
        if (wv < 2) __builtin_amdgcn_s_wait_tensorcnt(0);
        __syncthreads();

        // logits = q @ k^T : 4x4 tiles, K=32 -> one WMMA each
        for (int i = 0; i < 2; i++) {
            int id = wv * 2 + i;
            int mt = id >> 2, nt = id & 3;
            v8f acc = vzero8();
            v16h a = frag_a(&qs[(mt * 16 + ln16) * HD], h);
            v16h bf = frag_b(&ks[(nt * 16 + ln16) * HD], h);
            acc = wmma32(a, bf, acc);
            for (int j = 0; j < 8; j++)
                lg[(mt * 16 + j + 8 * h) * 64 + nt * 16 + ln16] = acc[j];
        }
        __syncthreads();

        // softmax rows with relative-position bias + shift mask
        if (tid < 64) {
            int i = tid;
            int ih = (i < MM) ? (i / WSZ) : 0, iw = (i < MM) ? (i % WSZ) : 0;
            int ci = (i < MM) ? cnt[i] : 0;
            float vals[MM];
            float mx = -3.0e38f;
            for (int j = 0; j < MM; j++) {
                int idx = (ih - j / WSZ + 6) * 13 + (iw - j % WSZ + 6);
                float v = lg[i * 64 + j] + btab[idx] + ((cnt[j] == ci) ? 0.0f : NEGV);
                vals[j] = v;
                mx = fmaxf(mx, v);
            }
            float s = 0.f;
            for (int j = 0; j < MM; j++) { float e = __expf(vals[j] - mx); vals[j] = e; s += e; }
            float inv = 1.0f / s;
            for (int j = 0; j < MM; j++) ps[i * 64 + j] = (_Float16)(vals[j] * inv);
            for (int j = MM; j < 64; j++) ps[i * 64 + j] = (_Float16)0.0f;
        }
        __syncthreads();

        // o = p @ v : 4x2 tiles, one per wave, K=64 -> 2 WMMA
        {
            int mt = wv >> 1, nt = wv & 1;
            v8f acc = vzero8();
            for (int k2 = 0; k2 < 2; k2++) {
                v16h a = frag_a(&ps[(mt * 16 + ln16) * 64 + k2 * 32], h);
                v16h bf = frag_b(&vt[(nt * 16 + ln16) * 64 + k2 * 32], h);
                acc = wmma32(a, bf, acc);
            }
            for (int j = 0; j < 8; j++)
                os[(mt * 16 + j + 8 * h) * DIM + head * HD + nt * 16 + ln16] = (_Float16)acc[j];
        }
        __syncthreads();
    }

    // out projection + bias + residual (in-place on xw): [64 x 192] = os * out_w^T
    for (int i = 0; i < 6; i++) {
        int id = wv * 6 + i;
        int mt = id / 12, nt = id % 12;
        v8f acc = vzero8();
        for (int ks2 = 0; ks2 < 6; ks2++) {
            v16h a = frag_a(&os[(mt * 16 + ln16) * DIM + ks2 * 32], h);
            v16h bf = frag_b(&outw16[(nt * 16 + ln16) * DIM + ks2 * 32], h);
            acc = wmma32(a, bf, acc);
        }
        int n = nt * 16 + ln16;
        for (int j = 0; j < 8; j++) {
            int m = mt * 16 + j + 8 * h;
            if (m < MM) {
                long idx = ((long)win * MM + m) * DIM + n;
                xw[idx] = xw[idx] + acc[j] + out_b[n];
            }
        }
    }
}

// ---------------- K4: LN2 + MLP (chunked hidden) + residual + window-reverse ----------------
__global__ __launch_bounds__(256) void k4_mlp(
    const float* __restrict__ xw, const float* __restrict__ ln2_g, const float* __restrict__ ln2_b,
    const _Float16* __restrict__ w116, const float* __restrict__ b1,
    const _Float16* __restrict__ w216, const float* __restrict__ b2,
    float* __restrict__ out) {
    __shared__ __align__(16) _Float16 y[64 * DIM];   // 24576 B
    __shared__ __align__(16) _Float16 hb[64 * 256];  // 32768 B
    int win = blockIdx.x;
    int tid = threadIdx.x, wv = tid >> 5, lane = tid & 31, ln16 = lane & 15, h = lane >> 4;

    if (tid < 64) {
        if (tid < MM) {
            const float* row = &xw[((long)win * MM + tid) * DIM];
            float s = 0.f, s2 = 0.f;
            for (int c = 0; c < DIM; c++) { float u = row[c]; s += u; s2 += u * u; }
            float m = s / DIM;
            float var = s2 / DIM - m * m;
            float inv = rsqrtf(var + 1e-5f);
            for (int c = 0; c < DIM; c++)
                y[tid * DIM + c] = (_Float16)((row[c] - m) * inv * ln2_g[c] + ln2_b[c]);
        } else {
            for (int c = 0; c < DIM; c++) y[tid * DIM + c] = (_Float16)0.0f;
        }
    }
    __syncthreads();

    v8f acc[6];
    for (int i = 0; i < 6; i++) acc[i] = vzero8();

    for (int chunk = 0; chunk < 3; chunk++) {
        // GEMM1: h_chunk[64 x 256] = y * w1^T (+b1, gelu)
        for (int i = 0; i < 8; i++) {
            int id = wv * 8 + i;
            int mt = id >> 4, nt = id & 15;
            int ng = chunk * 256 + nt * 16 + ln16;
            if (i + 1 < 8) {
                int ng2 = chunk * 256 + ((id + 1) & 15) * 16 + ln16;
                __builtin_prefetch(&w116[(long)ng2 * DIM], 0, 1);
            }
            v8f a1 = vzero8();
            for (int ks = 0; ks < 6; ks++) {
                v16h a = frag_a(&y[(mt * 16 + ln16) * DIM + ks * 32], h);
                v16h bf = frag_b(&w116[(long)ng * DIM + ks * 32], h);
                a1 = wmma32(a, bf, a1);
            }
            float bias = b1[ng];
            for (int j = 0; j < 8; j++) {
                float u = a1[j] + bias;
                float g = 0.5f * u * (1.0f + erff(u * 0.70710678118654752f));
                hb[(mt * 16 + j + 8 * h) * 256 + nt * 16 + ln16] = (_Float16)g;
            }
        }
        __syncthreads();
        // GEMM2 partial: acc += h_chunk * w2^T(chunk)
        for (int i = 0; i < 6; i++) {
            int id = wv * 6 + i;
            int mt = id / 12, nt = id % 12;
            for (int ks = 0; ks < 8; ks++) {
                v16h a = frag_a(&hb[(mt * 16 + ln16) * 256 + ks * 32], h);
                v16h bf = frag_b(&w216[(long)(nt * 16 + ln16) * HID + chunk * 256 + ks * 32], h);
                acc[i] = wmma32(a, bf, acc[i]);
            }
        }
        __syncthreads();
    }

    // epilogue: + b2 + residual, scatter with window-reverse + roll(+SS)
    int b = win >> 6, wi = win & 63, wh = wi >> 3, ww = wi & 7;
    for (int i = 0; i < 6; i++) {
        int id = wv * 6 + i;
        int mt = id / 12, nt = id % 12;
        int n = nt * 16 + ln16;
        for (int j = 0; j < 8; j++) {
            int m = mt * 16 + j + 8 * h;
            if (m < MM) {
                float val = acc[i][j] + b2[n] + xw[((long)win * MM + m) * DIM + n];
                int oh = (wh * WSZ + m / WSZ + SS) % IMG;
                int ow = (ww * WSZ + m % WSZ + SS) % IMG;
                out[(((long)b * DIM + n) * IMG + oh) * IMG + ow] = val;
            }
        }
    }
}

extern "C" void kernel_launch(void* const* d_in, const int* in_sizes, int n_in,
                              void* d_out, int out_size, void* d_ws, size_t ws_size,
                              hipStream_t stream) {
    const float* x      = (const float*)d_in[0];
    const float* ln1_g  = (const float*)d_in[1];
    const float* ln1_b  = (const float*)d_in[2];
    const float* qkv_w  = (const float*)d_in[3];
    const float* out_w  = (const float*)d_in[4];
    const float* out_b  = (const float*)d_in[5];
    const float* ln2_g  = (const float*)d_in[6];
    const float* ln2_b  = (const float*)d_in[7];
    const float* mlp_w1 = (const float*)d_in[8];
    const float* mlp_b1 = (const float*)d_in[9];
    const float* mlp_w2 = (const float*)d_in[10];
    const float* mlp_b2 = (const float*)d_in[11];
    const float* rel_t  = (const float*)d_in[12];
    float* out = (float*)d_out;

    char* ws = (char*)d_ws;
    const size_t sXW  = (size_t)NWIN * MM * DIM * sizeof(float);           // 154,140,672
    const size_t sQKV = (size_t)NWIN * HEADS * MM * HD * sizeof(_Float16); // 77,070,336 each
    float*    xw      = (float*)ws;
    _Float16* q16     = (_Float16*)(ws + sXW);
    _Float16* k16     = (_Float16*)(ws + sXW + sQKV);
    _Float16* v16     = (_Float16*)(ws + sXW + 2 * sQKV);
    char* wbase       = ws + sXW + 3 * sQKV;
    _Float16* qkvw16  = (_Float16*)(wbase);
    _Float16* outw16  = (_Float16*)(wbase + 576 * DIM * 2);
    _Float16* w116    = (_Float16*)(wbase + 576 * DIM * 2 + DIM * DIM * 2);
    _Float16* w216    = (_Float16*)(wbase + 576 * DIM * 2 + DIM * DIM * 2 + (size_t)HID * DIM * 2);

    k0_convert<<<512, 256, 0, stream>>>(qkv_w, out_w, mlp_w1, mlp_w2,
                                        qkvw16, outw16, w116, w216);
    k1_qkv<<<NWIN, 256, 0, stream>>>(x, ln1_g, ln1_b, qkvw16, xw, q16, k16, v16);
    k2_attn<<<NWIN, 256, 0, stream>>>(q16, k16, v16, outw16, out_b, rel_t, xw);
    k4_mlp<<<NWIN, 256, 0, stream>>>(xw, ln2_g, ln2_b, w116, mlp_b1, w216, mlp_b2, out);
}
